// mLSTMCell_45921790328939
// MI455X (gfx1250) — compile-verified, matrix operand order checked by reading, not run on hardware
//
#include <hip/hip_runtime.h>
#include <hip/hip_bf16.h>
#include <math.h>

// ---------------- types ----------------
typedef __attribute__((ext_vector_type(16))) __bf16 v16bf;
typedef __attribute__((ext_vector_type(8)))  float  v8f;
typedef __attribute__((ext_vector_type(4)))  float  f4;   // for nontemporal ld/st

#define D_H 512
#define D_IN 512
#define NB 256

// ---------------- fp32 -> bf16 (RNE) ----------------
__global__ void cvt_bf16_kernel(const float* __restrict__ src,
                                unsigned short* __restrict__ dst, int n) {
    int i = blockIdx.x * blockDim.x + threadIdx.x;
    if (i < n) {
        unsigned int u = __float_as_uint(src[i]);
        unsigned int r = (u + 0x7FFFu + ((u >> 16) & 1u)) >> 16;
        dst[i] = (unsigned short)r;
    }
}

// ---------------- scalar gates i_t, f_t ----------------
__global__ void gates_kernel(const float* __restrict__ x,
                             const float* __restrict__ wi_w, const float* __restrict__ wi_b,
                             const float* __restrict__ wf_w, const float* __restrict__ wf_b,
                             float* __restrict__ gi, float* __restrict__ gf) {
    __shared__ float ri[4];
    __shared__ float rf[4];
    const int b    = blockIdx.x;
    const int t    = threadIdx.x;                  // 128 threads, float4 each
    const int wave = t >> 5, lane = t & 31;
    const float4* xr  = (const float4*)(x + (size_t)b * D_IN);
    const float4* wiv = (const float4*)wi_w;
    const float4* wfv = (const float4*)wf_w;
    float4 xv = xr[t], a = wiv[t], c = wfv[t];
    float pi = xv.x * a.x + xv.y * a.y + xv.z * a.z + xv.w * a.w;
    float pf = xv.x * c.x + xv.y * c.y + xv.z * c.z + xv.w * c.w;
    #pragma unroll
    for (int off = 16; off > 0; off >>= 1) {
        pi += __shfl_xor(pi, off, 32);
        pf += __shfl_xor(pf, off, 32);
    }
    if (lane == 0) { ri[wave] = pi; rf[wave] = pf; }
    __syncthreads();
    if (t == 0) {
        float si = ri[0] + ri[1] + ri[2] + ri[3];
        float sf = rf[0] + rf[1] + rf[2] + rf[3];
        gi[b] = fminf(expf(si + wi_b[0]), 50.0f);
        gf[b] = 1.0f / (1.0f + expf(-(sf + wf_b[0])));
    }
}

// ---------------- WMMA projection GEMM: Y[m,n] = sum_k X[m,k] * W[n,k] ----------------
// grid = (32 n-tiles, 16 m-tiles, 4 matrices), block = 32 (one wave per 16x16 tile)
__global__ void proj_wmma_kernel(const unsigned short* __restrict__ xb,
                                 const unsigned short* __restrict__ wb,
                                 float* __restrict__ yout) {
    const int lane = threadIdx.x;          // 0..31
    const int half = lane >> 4;            // lane half selects K sub-range
    const int l16  = lane & 15;
    const int tn = blockIdx.x;             // N tile
    const int tm = blockIdx.y;             // M tile
    const int w  = blockIdx.z;             // which weight matrix

    const unsigned short* Wm = wb + (size_t)w * D_H * D_IN;
    float* Y = yout + (size_t)w * NB * D_H;

    const int rowA = tm * 16 + l16;        // A: one M-row per lane (16-bit A layout)
    const int colB = tn * 16 + l16;        // B: one N-col per lane = row of W (contiguous)

    const unsigned int* arow = (const unsigned int*)(xb + (size_t)rowA * D_IN);
    const unsigned int* brow = (const unsigned int*)(Wm + (size_t)colB * D_IN);

    union Frag { v16bf v; unsigned int u[8]; };
    v8f acc = {};

    #pragma unroll
    for (int kk = 0; kk < D_IN; kk += 32) {
        Frag a, b;
        #pragma unroll
        for (int r = 0; r < 8; ++r) {
            // 16-bit A 16x32 layout: VGPR r holds a K-pair; lanes 16-31 offset by 8;
            // VGPRs 4..7 cover K=16..31.
            int k0 = kk + ((r < 4) ? 0 : 16) + half * 8 + (r & 3) * 2;
            a.u[r] = arow[k0 >> 1];
            b.u[r] = brow[k0 >> 1];
        }
        acc = __builtin_amdgcn_wmma_f32_16x16x32_bf16(
            /*neg_a=*/false, a.v, /*neg_b=*/false, b.v,
            /*c_mod=*/(short)0, acc, /*reuse_a=*/false, /*reuse_b=*/false);
    }

    #pragma unroll
    for (int r = 0; r < 8; ++r) {          // D layout: VGPR r -> M = r (+8 for hi lanes)
        int m = r + half * 8;
        Y[(size_t)(tm * 16 + m) * D_H + tn * 16 + l16] = acc[r];
    }
}

// ---------------- bandwidth-dominant stream: C_t = f*C_prev + i*v*k^T, h_acc = C_t q ----
// grid = NB*128 blocks, 128 threads, 4 rows (512 floats each) per block.
// C stream (536 MB) >> L2 (192 MB): use non-temporal loads/stores so the stream
// doesn't evict the reusable small tensors; wave32 shfl reductions, one barrier.
__global__ void stream_update_kernel(const float* __restrict__ Cprev,
                                     const float* __restrict__ qv,
                                     const float* __restrict__ kv,
                                     const float* __restrict__ vv,
                                     const float* __restrict__ gi,
                                     const float* __restrict__ gf,
                                     float* __restrict__ Cout,
                                     float* __restrict__ hacc) {
    __shared__ float red[4][4];                  // [row][wave]
    const int t    = threadIdx.x;                // 0..127, float4 each -> 512 floats/row
    const int wave = t >> 5, lane = t & 31;
    const int blk  = blockIdx.x;
    const int b    = blk >> 7;                   // 128 blocks per batch
    const int i0   = (blk & 127) * 4;            // 4 rows per block
    const float f  = gf[b];
    const float it = gi[b];

    const f4 q4 = ((const f4*)(qv + (size_t)b * D_H))[t];
    const f4 k4 = ((const f4*)(kv + (size_t)b * D_H))[t];

    #pragma unroll
    for (int r = 0; r < 4; ++r) {
        const int i = i0 + r;
        const f4* crow = (const f4*)(Cprev + ((size_t)b * D_H + i) * D_H);
        f4*       orow = (f4*)(Cout  + ((size_t)b * D_H + i) * D_H);
        if (r < 3) __builtin_prefetch(crow + 128 + t, 0, 0);   // global_prefetch next row
        f4 c = __builtin_nontemporal_load(crow + t);           // TH=NT streaming read
        const float s = it * vv[(size_t)b * D_H + i];
        f4 o;
        o.x = f * c.x + s * k4.x;
        o.y = f * c.y + s * k4.y;
        o.z = f * c.z + s * k4.z;
        o.w = f * c.w + s * k4.w;
        __builtin_nontemporal_store(o, orow + t);              // TH=NT streaming write
        float p = o.x * q4.x + o.y * q4.y + o.z * q4.z + o.w * q4.w;
        #pragma unroll
        for (int off = 16; off > 0; off >>= 1)                 // wave32 butterfly
            p += __shfl_xor(p, off, 32);
        if (lane == 0) red[r][wave] = p;
    }
    __syncthreads();
    if (t < 4) {                                               // one thread per row
        float s = red[t][0] + red[t][1] + red[t][2] + red[t][3];
        hacc[(size_t)b * D_H + i0 + t] = s;
    }
}

// ---------------- epilogue: n_t, nq, output gate, LayerNorm ----------------
// grid = NB blocks, 256 threads, each handles j and j+256
__global__ void epilogue_kernel(const float* __restrict__ qv,
                                const float* __restrict__ kv,
                                const float* __restrict__ opre,
                                const float* __restrict__ Wo_b,
                                const float* __restrict__ n_prev,
                                const float* __restrict__ gi,
                                const float* __restrict__ gf,
                                const float* __restrict__ hacc,
                                const float* __restrict__ ln_g,
                                const float* __restrict__ ln_b,
                                float* __restrict__ out_h,
                                float* __restrict__ out_n) {
    __shared__ float red[8];
    __shared__ float sval;
    const int b    = blockIdx.x;
    const int t    = threadIdx.x;
    const int wave = t >> 5, lane = t & 31;
    const size_t base = (size_t)b * D_H;
    const int j0 = t, j1 = t + 256;
    const float f = gf[b], it = gi[b];

    const float q0 = qv[base + j0], q1 = qv[base + j1];
    const float n0 = f * n_prev[base + j0] + it * kv[base + j0];
    const float n1 = f * n_prev[base + j1] + it * kv[base + j1];
    out_n[base + j0] = n0;
    out_n[base + j1] = n1;

    // --- reduction helper pattern: shfl butterfly + 8-wave LDS combine ---
    // nq = dot(n_t, q)
    float p = n0 * q0 + n1 * q1;
    #pragma unroll
    for (int off = 16; off > 0; off >>= 1) p += __shfl_xor(p, off, 32);
    if (lane == 0) red[wave] = p;
    __syncthreads();
    if (t == 0) {
        float s = 0.f;
        #pragma unroll
        for (int wv = 0; wv < 8; ++wv) s += red[wv];
        sval = s;
    }
    __syncthreads();
    const float denom = fmaxf(fabsf(sval), 1.0f);

    const float o0 = 1.0f / (1.0f + expf(-(opre[base + j0] + Wo_b[j0])));
    const float o1 = 1.0f / (1.0f + expf(-(opre[base + j1] + Wo_b[j1])));
    const float h0 = o0 * hacc[base + j0] / denom;
    const float h1 = o1 * hacc[base + j1] / denom;

    // mean
    p = h0 + h1;
    #pragma unroll
    for (int off = 16; off > 0; off >>= 1) p += __shfl_xor(p, off, 32);
    if (lane == 0) red[wave] = p;
    __syncthreads();
    if (t == 0) {
        float s = 0.f;
        #pragma unroll
        for (int wv = 0; wv < 8; ++wv) s += red[wv];
        sval = s * (1.0f / (float)D_H);
    }
    __syncthreads();
    const float mu = sval;
    const float d0 = h0 - mu, d1 = h1 - mu;

    // variance
    p = d0 * d0 + d1 * d1;
    #pragma unroll
    for (int off = 16; off > 0; off >>= 1) p += __shfl_xor(p, off, 32);
    if (lane == 0) red[wave] = p;
    __syncthreads();
    if (t == 0) {
        float s = 0.f;
        #pragma unroll
        for (int wv = 0; wv < 8; ++wv) s += red[wv];
        sval = s * (1.0f / (float)D_H);
    }
    __syncthreads();
    const float inv = rsqrtf(sval + 1e-5f);

    out_h[base + j0] = d0 * inv * ln_g[j0] + ln_b[j0];
    out_h[base + j1] = d1 * inv * ln_g[j1] + ln_b[j1];
}

// ---------------- launch ----------------
extern "C" void kernel_launch(void* const* d_in, const int* in_sizes, int n_in,
                              void* d_out, int out_size, void* d_ws, size_t ws_size,
                              hipStream_t stream) {
    const float* x      = (const float*)d_in[0];
    const float* C_prev = (const float*)d_in[1];
    const float* n_prev = (const float*)d_in[2];
    const float* Wq     = (const float*)d_in[3];
    const float* Wk     = (const float*)d_in[4];
    const float* Wv     = (const float*)d_in[5];
    const float* wi_w   = (const float*)d_in[6];
    const float* wi_b   = (const float*)d_in[7];
    const float* wf_w   = (const float*)d_in[8];
    const float* wf_b   = (const float*)d_in[9];
    const float* Wo_w   = (const float*)d_in[10];
    const float* Wo_b   = (const float*)d_in[11];
    const float* ln_g   = (const float*)d_in[12];
    const float* ln_b   = (const float*)d_in[13];

    float* out_h = (float*)d_out;                                // [256,512]
    float* out_C = out_h + (size_t)NB * D_H;                     // [256,512,512]
    float* out_n = out_C + (size_t)NB * D_H * D_H;               // [256,512]

    // workspace layout (~4.8 MB)
    char* ws = (char*)d_ws;
    unsigned short* xb   = (unsigned short*)(ws);                          // 256*512 bf16
    unsigned short* Wb   = (unsigned short*)(ws + 262144);                 // 4 * 512*512 bf16
    float*          qkvo = (float*)(ws + 262144 + 2097152);                // 4 * 256*512 f32
    float*          gi   = (float*)(ws + 4456448);                         // 256
    float*          gf   = gi + 256;
    float*          hacc = (float*)(ws + 4458496);                         // 256*512

    const int NX = NB * D_IN;        // 131072
    const int NW = D_H * D_IN;       // 262144

    // 1) bf16 conversions
    cvt_bf16_kernel<<<(NX + 255) / 256, 256, 0, stream>>>(x,    xb,               NX);
    cvt_bf16_kernel<<<(NW + 255) / 256, 256, 0, stream>>>(Wq,   Wb + 0*(size_t)NW, NW);
    cvt_bf16_kernel<<<(NW + 255) / 256, 256, 0, stream>>>(Wk,   Wb + 1*(size_t)NW, NW);
    cvt_bf16_kernel<<<(NW + 255) / 256, 256, 0, stream>>>(Wv,   Wb + 2*(size_t)NW, NW);
    cvt_bf16_kernel<<<(NW + 255) / 256, 256, 0, stream>>>(Wo_w, Wb + 3*(size_t)NW, NW);

    // 2) scalar gates
    gates_kernel<<<NB, 128, 0, stream>>>(x, wi_w, wi_b, wf_w, wf_b, gi, gf);

    // 3) WMMA projections: q,k,v,o_pre (one wave per 16x16 tile)
    proj_wmma_kernel<<<dim3(D_H / 16, NB / 16, 4), 32, 0, stream>>>(xb, Wb, qkvo);

    float* qv   = qkvo + 0 * (size_t)NB * D_H;
    float* kv   = qkvo + 1 * (size_t)NB * D_H;
    float* vv   = qkvo + 2 * (size_t)NB * D_H;
    float* opre = qkvo + 3 * (size_t)NB * D_H;

    // 4) streaming C update + fused retrieval dot (bandwidth-bound: ~536 MB, NT hints)
    stream_update_kernel<<<NB * 128, 128, 0, stream>>>(C_prev, qv, kv, vv, gi, gf,
                                                       out_C, hacc);

    // 5) epilogue: n_t, normalization, gate, LayerNorm
    epilogue_kernel<<<NB, 256, 0, stream>>>(qv, kv, opre, Wo_b, n_prev, gi, gf,
                                            hacc, ln_g, ln_b, out_h, out_n);
}
